// PoolingModule_27616639713602
// MI455X (gfx1250) — matrix operand order repaired and checked by reading, not compiled, not run on hardware
//
#include <hip/hip_runtime.h>
#include <hip/hip_bf16.h>

typedef __bf16 bf16_t;
typedef __attribute__((ext_vector_type(16))) __bf16 v16bf;
typedef __attribute__((ext_vector_type(8)))  __bf16 v8bf;
typedef __attribute__((ext_vector_type(8)))  float  v8f;
typedef __attribute__((ext_vector_type(4)))  unsigned int u32x4;
typedef __attribute__((ext_vector_type(8)))  int i32x8;
typedef __attribute__((ext_vector_type(4)))  int i32x4;

#define D_DIM 1024
#define BM 128
#define BK 32
#define LDAB 40      // padded stride for 128x32 staging tiles (16B aligned, bank-spread)
#define LDKV 72      // padded stride for 64-wide tiles
#define LDBIG 1032   // padded stride for 128x1024 resident tiles
#define NTHREADS 256
#define SCALE_Q 0.125f   // R^-0.5, R=64
#define INV_S (1.0f/4096.0f)

// ---------- bf16 helpers: native hardware converts (RNE) ----------
__device__ __forceinline__ bf16_t f2bf(float f) { return (bf16_t)f; }
__device__ __forceinline__ float bf2f(bf16_t b) { return (float)b; }

// ---------- WMMA wrapper ----------
__device__ __forceinline__ v8f wmma_bf16(v16bf a, v16bf b, v8f c) {
  return __builtin_amdgcn_wmma_f32_16x16x32_bf16(false, a, false, b, (short)0, c, false, false);
}

// A fragment: rows m0..m0+15 (M), K window [k0,k0+32)
__device__ __forceinline__ v16bf load_a_frag(const bf16_t* base, int ld, int m0, int k0, int lane) {
  int hl = lane >> 4;
  const bf16_t* p = base + (m0 + (lane & 15)) * ld + k0 + hl * 8;
  union { v16bf v; v8bf h[2]; } u;
  u.h[0] = *(const v8bf*)p;
  u.h[1] = *(const v8bf*)(p + 16);
  return u.v;
}
// B fragment from N-major stored B: tile t covers N = 16t..16t+15
__device__ __forceinline__ v16bf load_b_frag(const bf16_t* base, int ld, int t, int k0, int lane) {
  int hl = lane >> 4;
  const bf16_t* p = base + (t * 16 + (lane & 15)) * ld + k0 + hl * 8;
  union { v16bf v; v8bf h[2]; } u;
  u.h[0] = *(const v8bf*)p;
  u.h[1] = *(const v8bf*)(p + 16);
  return u.v;
}

// ---------- Tensor Data Mover: 2D tile load, global(bf16) -> LDS, with LDS row padding ----------
// D# packing per CDNA5 ISA (group0: count/lds/global/type,
// group1: data_size=2B, pad_enable, pad codes, tensor dims, tile dims, dim0 stride).
__device__ __forceinline__ unsigned lds_of(const void* p) {
  return (unsigned)(unsigned long long)p;   // LDS aperture: addr[31:0] = LDS byte offset
}
__device__ __forceinline__ void tdm_load_2d(unsigned lds_addr, const void* gaddr,
                                            unsigned tensor_w, unsigned tensor_h,
                                            unsigned tile_w, unsigned tile_h,
                                            unsigned pad_interval, unsigned pad_amount) {
  unsigned long long ga = (unsigned long long)gaddr;
  u32x4 g0;
  g0[0] = 1u;                                              // count=1, user mode
  g0[1] = lds_addr;
  g0[2] = (unsigned)(ga & 0xFFFFFFFFu);
  g0[3] = ((unsigned)(ga >> 32) & 0x01FFFFFFu) | (2u << 30); // addr hi | type=2 (image)
  i32x8 g1;
  g1[0] = (int)((1u << 16)                                 // data_size = 1 -> 2 bytes
              | (1u << 20)                                 // pad_enable
              | (pad_interval << 22) | (pad_amount << 25));
  g1[1] = (int)((tensor_w & 0xFFFFu) << 16);               // tensor_dim0[15:0]
  g1[2] = (int)((tensor_w >> 16) | ((tensor_h & 0xFFFFu) << 16));
  g1[3] = (int)((tensor_h >> 16) | (tile_w << 16));        // tensor_dim1 hi | tile_dim0
  g1[4] = (int)(tile_h & 0xFFFFu);                         // tile_dim1 (tile_dim2 = 0)
  g1[5] = (int)tensor_w;                                   // tensor_dim0_stride[31:0]
  g1[6] = 0;
  g1[7] = 0;
  i32x4 z4 = {0, 0, 0, 0};
  i32x8 z8 = {0, 0, 0, 0, 0, 0, 0, 0};
  __builtin_amdgcn_tensor_load_to_lds(g0, g1, z4, z4, z8, 0);
}
// pad codes: rows of 32 elems (64B = 16 DW) + 8 elem pad (4 DW) -> (3,3); rows of 64 elems -> (4,3)
#define PADI_32 3
#define PADI_64 4
#define PADA_4DW 3

// ---------- cooperative f32 -> bf16 tile copy (x only), packed converts ----------
__device__ __forceinline__ void copy_tile_f32_to_bf16(bf16_t* dst, int dstld, const float* src,
                                                      long srcld, int rows, int cols, int tid) {
  int cpr = cols >> 3;
  int chunks = rows * cpr;
  for (int c = tid; c < chunks; c += NTHREADS) {
    int r = c / cpr;
    int co = (c - r * cpr) << 3;
    v8f f = *(const v8f*)(src + (long)r * srcld + co);
    *(v8bf*)(dst + r * dstld + co) = __builtin_convertvector(f, v8bf);
  }
}

// ---------- prep kernels ----------
__global__ void k_zero(float* p, int n) {
  int i = blockIdx.x * blockDim.x + threadIdx.x;
  if (i < n) p[i] = 0.0f;
}

__global__ void k_wcvt(const float* __restrict__ wq, const float* __restrict__ wo,
                       const float* __restrict__ w1, const float* __restrict__ w2,
                       bf16_t* __restrict__ dst) {
  long gid = (long)blockIdx.x * NTHREADS + threadIdx.x;     // 4 * 1M total
  int which = (int)(gid >> 20);
  long idx = gid & ((1l << 20) - 1);
  int n = (int)(idx >> 10), k = (int)(idx & 1023);
  const float* src = (which == 0) ? wq : (which == 1) ? wo : (which == 2) ? w1 : w2;
  dst[gid] = f2bf(src[(long)k * D_DIM + n]);
}

__global__ void k_kvcvt(const float* __restrict__ kv, bf16_t* __restrict__ kvbf,
                        bf16_t* __restrict__ kvT) {
  int gid = blockIdx.x * NTHREADS + threadIdx.x;            // 65536
  kvbf[gid] = f2bf(kv[gid]);
  int h = gid >> 12, rem = gid & 4095, dd = rem >> 6, r = rem & 63;
  kvT[gid] = f2bf(kv[(long)r * D_DIM + h * 64 + dd]);
}

// ---------- fused attention front-end + OUT = AT @ woT ----------
#define SMEM_ATTN (BM*LDBIG*2 + 20480 + 64*LDKV*2*2 + BM*LDKV*2)
__global__ void __launch_bounds__(NTHREADS)
k_attn(const float* __restrict__ x, const bf16_t* __restrict__ wqT,
       const bf16_t* __restrict__ woT, const bf16_t* __restrict__ kvbf,
       const bf16_t* __restrict__ kvT, bf16_t* __restrict__ outb) {
  extern __shared__ __align__(16) char smem[];
  bf16_t* sAT  = (bf16_t*)smem;                               // [128][1032]
  char*   abr  = smem + BM * LDBIG * 2;
  bf16_t* sA   = (bf16_t*)abr;                                // [128][40]
  bf16_t* sB   = (bf16_t*)(abr + BM * LDAB * 2);              // [128][40]
  bf16_t* sXQ  = (bf16_t*)abr;                                // [128][72] overlays sA/sB
  bf16_t* sKV  = (bf16_t*)(abr + 20480);                      // [64][72]
  bf16_t* sKVT = sKV + 64 * LDKV;                             // [64][72]
  bf16_t* sP   = sKVT + 64 * LDKV;                            // [128][72]

  const int tid  = threadIdx.x;
  const int lane = tid & 31;
  const int wv   = tid >> 5;
  const int m0   = wv * 16;
  const int hl   = lane >> 4;
  const int cl   = lane & 15;
  const long tileM = (long)blockIdx.x * BM;

  #pragma unroll 1
  for (int nb = 0; nb < 8; ++nb) {
    const bf16_t* wqblk = wqT + (long)(nb * 128) * D_DIM;
    // ---- GEMM1: xq block [128 x 128] = x[128 x 1024] @ wqT-block ----
    v8f acc[8];
    #pragma unroll
    for (int t = 0; t < 8; ++t) acc[t] = (v8f){0,0,0,0,0,0,0,0};
    #pragma unroll 1
    for (int kk = 0; kk < D_DIM; kk += BK) {
      __syncthreads();
      if (wv == 0)
        tdm_load_2d(lds_of(sB), wqblk + kk, D_DIM, D_DIM, BK, BM, PADI_32, PADA_4DW);
      if (kk + BK < D_DIM)
        __builtin_prefetch(x + tileM * D_DIM + kk + BK, 0, 1);
      copy_tile_f32_to_bf16(sA, LDAB, x + tileM * D_DIM + kk, D_DIM, BM, BK, tid);
      if (wv == 0) __builtin_amdgcn_s_wait_tensorcnt(0);
      __syncthreads();
      v16bf a = load_a_frag(sA, LDAB, m0, 0, lane);
      #pragma unroll
      for (int t = 0; t < 8; ++t)
        acc[t] = wmma_bf16(a, load_b_frag(sB, LDAB, t, 0, lane), acc[t]);
    }
    // ---- attention for the 2 heads of this block ----
    #pragma unroll 1
    for (int hh = 0; hh < 2; ++hh) {
      const int head = nb * 2 + hh;
      __syncthreads();   // sA/sB readers (and prev head's sXQ readers) done
      #pragma unroll
      for (int t2 = 0; t2 < 4; ++t2) {
        int t = hh * 4 + t2;
        #pragma unroll
        for (int v = 0; v < 8; ++v)
          sXQ[(m0 + v + hl * 8) * LDKV + t2 * 16 + cl] = f2bf(acc[t][v] * SCALE_Q);
      }
      if (wv == 0) {
        tdm_load_2d(lds_of(sKV), kvbf + head * 64, D_DIM, 64, 64, 64, PADI_64, PADA_4DW);
        tdm_load_2d(lds_of(sKVT), kvT + (long)head * 4096, 64, 1024, 64, 64, PADI_64, PADA_4DW);
        __builtin_amdgcn_s_wait_tensorcnt(0);
      }
      __syncthreads();
      // scores [16 x 64] per wave = xq_h @ kv_h^T   (K = hd = 64)
      v8f sc[4];
      #pragma unroll
      for (int t = 0; t < 4; ++t) sc[t] = (v8f){0,0,0,0,0,0,0,0};
      #pragma unroll
      for (int k2 = 0; k2 < 64; k2 += 32) {
        v16bf a = load_a_frag(sXQ, LDKV, m0, k2, lane);
        #pragma unroll
        for (int t = 0; t < 4; ++t)
          sc[t] = wmma_bf16(a, load_b_frag(sKV, LDKV, t, k2, lane), sc[t]);
      }
      // softmax over 64 latents (16-lane half-wave reductions match C layout)
      #pragma unroll
      for (int v = 0; v < 8; ++v) {
        float m = fmaxf(fmaxf(sc[0][v], sc[1][v]), fmaxf(sc[2][v], sc[3][v]));
        #pragma unroll
        for (int off = 8; off >= 1; off >>= 1) m = fmaxf(m, __shfl_xor(m, off, 32));
        float e0 = __expf(sc[0][v] - m), e1 = __expf(sc[1][v] - m);
        float e2 = __expf(sc[2][v] - m), e3 = __expf(sc[3][v] - m);
        float sum = e0 + e1 + e2 + e3;
        #pragma unroll
        for (int off = 8; off >= 1; off >>= 1) sum += __shfl_xor(sum, off, 32);
        float inv = 1.0f / sum;
        sc[0][v] = e0 * inv; sc[1][v] = e1 * inv; sc[2][v] = e2 * inv; sc[3][v] = e3 * inv;
      }
      #pragma unroll
      for (int t = 0; t < 4; ++t)
        #pragma unroll
        for (int v = 0; v < 8; ++v)
          sP[(m0 + v + hl * 8) * LDKV + t * 16 + cl] = f2bf(sc[t][v]);
      __syncthreads();
      // out_h [16 x 64] per wave = P @ kv_h   (K = R = 64)
      v8f o[4];
      #pragma unroll
      for (int t = 0; t < 4; ++t) o[t] = (v8f){0,0,0,0,0,0,0,0};
      #pragma unroll
      for (int k2 = 0; k2 < 64; k2 += 32) {
        v16bf a = load_a_frag(sP, LDKV, m0, k2, lane);
        #pragma unroll
        for (int t = 0; t < 4; ++t)
          o[t] = wmma_bf16(a, load_b_frag(sKVT, LDKV, t, k2, lane), o[t]);
      }
      #pragma unroll
      for (int t = 0; t < 4; ++t)
        #pragma unroll
        for (int v = 0; v < 8; ++v)
          sAT[(m0 + v + hl * 8) * LDBIG + nb * 128 + hh * 64 + t * 16 + cl] = f2bf(o[t][v]);
    }
  }

  // ---- OUT = AT @ woT, AT resident in LDS; woT tiles double-buffered via TDM ----
  bf16_t* b0 = (bf16_t*)abr;
  bf16_t* b1 = (bf16_t*)(abr + BM * LDAB * 2);
  #pragma unroll 1
  for (int nb = 0; nb < 8; ++nb) {
    const bf16_t* woblk = woT + (long)(nb * 128) * D_DIM;
    v8f acc[8];
    #pragma unroll
    for (int t = 0; t < 8; ++t) acc[t] = (v8f){0,0,0,0,0,0,0,0};
    __syncthreads();
    if (wv == 0)
      tdm_load_2d(lds_of(b0), woblk, D_DIM, D_DIM, BK, BM, PADI_32, PADA_4DW);
    #pragma unroll 1
    for (int ks = 0; ks < 32; ++ks) {
      bf16_t* bcur = (ks & 1) ? b1 : b0;
      bf16_t* bnxt = (ks & 1) ? b0 : b1;
      if (wv == 0) __builtin_amdgcn_s_wait_tensorcnt(0);
      __syncthreads();
      if (wv == 0 && ks + 1 < 32)
        tdm_load_2d(lds_of(bnxt), woblk + (ks + 1) * BK, D_DIM, D_DIM, BK, BM, PADI_32, PADA_4DW);
      v16bf a = load_a_frag(sAT, LDBIG, m0, ks * BK, lane);
      #pragma unroll
      for (int t = 0; t < 8; ++t)
        acc[t] = wmma_bf16(a, load_b_frag(bcur, LDAB, t, 0, lane), acc[t]);
    }
    #pragma unroll
    for (int t = 0; t < 8; ++t)
      #pragma unroll
      for (int v = 0; v < 8; ++v) {
        long row = tileM + m0 + v + hl * 8;
        int col = nb * 128 + t * 16 + cl;
        outb[row * D_DIM + col] = f2bf(acc[t][v]);
      }
  }
}

// ---------- fused MLP + residual + segment mean-pool (all tiles via TDM, double-buffered) ----------
#define SMEM_MLP (BM*LDBIG*2 + 4*BM*LDAB*2)
__global__ void __launch_bounds__(NTHREADS)
k_mlp(const bf16_t* __restrict__ outb, const bf16_t* __restrict__ w1T,
      const bf16_t* __restrict__ w2T, const float* __restrict__ b1,
      const float* __restrict__ b2, float* __restrict__ dout) {
  extern __shared__ __align__(16) char smem[];
  bf16_t* sH  = (bf16_t*)smem;                                // [128][1032]
  char*   abr = smem + BM * LDBIG * 2;
  bf16_t* sA0 = (bf16_t*)abr;                                 // [128][40] x4
  bf16_t* sA1 = sA0 + BM * LDAB;
  bf16_t* sB0 = sA1 + BM * LDAB;
  bf16_t* sB1 = sB0 + BM * LDAB;

  const int tid  = threadIdx.x;
  const int lane = tid & 31;
  const int wv   = tid >> 5;
  const int m0   = wv * 16;
  const int hl   = lane >> 4;
  const int cl   = lane & 15;
  const long tileM = (long)blockIdx.x * BM;
  const int batch = blockIdx.x >> 5;          // 32 tiles of 128 tokens per batch (S = 4096)
  const long NROWS = 131072;

  // ---- stage 1: h = gelu(OUT @ w1T + b1), resident in LDS ----
  #pragma unroll 1
  for (int nb = 0; nb < 8; ++nb) {
    const bf16_t* w1blk = w1T + (long)(nb * 128) * D_DIM;
    v8f acc[8];
    #pragma unroll
    for (int t = 0; t < 8; ++t) acc[t] = (v8f){0,0,0,0,0,0,0,0};
    __syncthreads();
    if (wv == 0) {
      tdm_load_2d(lds_of(sA0), outb + tileM * D_DIM, D_DIM, NROWS, BK, BM, PADI_32, PADA_4DW);
      tdm_load_2d(lds_of(sB0), w1blk, D_DIM, D_DIM, BK, BM, PADI_32, PADA_4DW);
    }
    #pragma unroll 1
    for (int ks = 0; ks < 32; ++ks) {
      bf16_t* acur = (ks & 1) ? sA1 : sA0;
      bf16_t* anxt = (ks & 1) ? sA0 : sA1;
      bf16_t* bcur = (ks & 1) ? sB1 : sB0;
      bf16_t* bnxt = (ks & 1) ? sB0 : sB1;
      if (wv == 0) __builtin_amdgcn_s_wait_tensorcnt(0);
      __syncthreads();
      if (wv == 0 && ks + 1 < 32) {
        tdm_load_2d(lds_of(anxt), outb + tileM * D_DIM + (ks + 1) * BK, D_DIM, NROWS, BK, BM, PADI_32, PADA_4DW);
        tdm_load_2d(lds_of(bnxt), w1blk + (ks + 1) * BK, D_DIM, D_DIM, BK, BM, PADI_32, PADA_4DW);
      }
      v16bf a = load_a_frag(acur, LDAB, m0, 0, lane);
      #pragma unroll
      for (int t = 0; t < 8; ++t)
        acc[t] = wmma_bf16(a, load_b_frag(bcur, LDAB, t, 0, lane), acc[t]);
    }
    #pragma unroll
    for (int t = 0; t < 8; ++t) {
      int col = nb * 128 + t * 16 + cl;
      float bias = b1[col];
      #pragma unroll
      for (int v = 0; v < 8; ++v) {
        float xv = acc[t][v] + bias;
        float g = 0.5f * xv * (1.0f + erff(xv * 0.70710678118654752f));
        sH[(m0 + v + hl * 8) * LDBIG + col] = f2bf(g);
      }
    }
  }

  // ---- stage 2: y = h @ w2T + b2 + OUT ; mean-pool into d_out ----
  #pragma unroll 1
  for (int nb = 0; nb < 8; ++nb) {
    const bf16_t* w2blk = w2T + (long)(nb * 128) * D_DIM;
    v8f acc[8];
    #pragma unroll
    for (int t = 0; t < 8; ++t) acc[t] = (v8f){0,0,0,0,0,0,0,0};
    __syncthreads();
    if (wv == 0)
      tdm_load_2d(lds_of(sB0), w2blk, D_DIM, D_DIM, BK, BM, PADI_32, PADA_4DW);
    #pragma unroll 1
    for (int ks = 0; ks < 32; ++ks) {
      bf16_t* bcur = (ks & 1) ? sB1 : sB0;
      bf16_t* bnxt = (ks & 1) ? sB0 : sB1;
      if (wv == 0) __builtin_amdgcn_s_wait_tensorcnt(0);
      __syncthreads();
      if (wv == 0 && ks + 1 < 32)
        tdm_load_2d(lds_of(bnxt), w2blk + (ks + 1) * BK, D_DIM, D_DIM, BK, BM, PADI_32, PADA_4DW);
      v16bf a = load_a_frag(sH, LDBIG, m0, ks * BK, lane);
      #pragma unroll
      for (int t = 0; t < 8; ++t)
        acc[t] = wmma_bf16(a, load_b_frag(bcur, LDAB, t, 0, lane), acc[t]);
    }
    #pragma unroll
    for (int t = 0; t < 8; ++t) {
      int col = nb * 128 + t * 16 + cl;
      float bias = b2[col];
      float psum = 0.0f;
      #pragma unroll
      for (int v = 0; v < 8; ++v) {
        long row = tileM + m0 + v + hl * 8;
        float y = acc[t][v] + bias + bf2f(outb[row * D_DIM + col]);
        psum += y;
      }
      psum += __shfl_xor(psum, 16, 32);   // fold the two lane-halves (rows m0+8..m0+15)
      if (lane < 16) atomicAdd(&dout[batch * D_DIM + col], psum * INV_S);
    }
  }
}

// ---------- host launcher ----------
extern "C" void kernel_launch(void* const* d_in, const int* in_sizes, int n_in,
                              void* d_out, int out_size, void* d_ws, size_t ws_size,
                              hipStream_t stream) {
  const float* x  = (const float*)d_in[0];
  const float* wq = (const float*)d_in[1];
  const float* kv = (const float*)d_in[2];
  const float* wo = (const float*)d_in[3];
  const float* w1 = (const float*)d_in[4];
  const float* b1 = (const float*)d_in[5];
  const float* w2 = (const float*)d_in[6];
  const float* b2 = (const float*)d_in[7];
  float* out = (float*)d_out;

  char* ws = (char*)d_ws;
  const size_t MB = 1024ull * 1024ull;
  bf16_t* wqT  = (bf16_t*)(ws + 0 * MB);     // 2 MB each, N-major
  bf16_t* woT  = (bf16_t*)(ws + 2 * MB);
  bf16_t* w1T  = (bf16_t*)(ws + 4 * MB);
  bf16_t* w2T  = (bf16_t*)(ws + 6 * MB);
  bf16_t* kvbf = (bf16_t*)(ws + 8 * MB);                 // [64][1024]
  bf16_t* kvT  = (bf16_t*)(ws + 8 * MB + 131072);        // [16][64][64]
  bf16_t* outb = (bf16_t*)(ws + 8 * MB + 262144);        // [131072][1024] bf16 = 256 MB

  k_wcvt<<<(4 * 1024 * 1024) / NTHREADS, NTHREADS, 0, stream>>>(wq, wo, w1, w2, wqT);
  k_kvcvt<<<65536 / NTHREADS, NTHREADS, 0, stream>>>(kv, kvbf, kvT);
  k_zero<<<(out_size + NTHREADS - 1) / NTHREADS, NTHREADS, 0, stream>>>(out, out_size);

  (void)hipFuncSetAttribute((const void*)k_attn, hipFuncAttributeMaxDynamicSharedMemorySize, SMEM_ATTN);
  k_attn<<<1024, NTHREADS, SMEM_ATTN, stream>>>(x, wqT, woT, kvbf, kvT, outb);

  (void)hipFuncSetAttribute((const void*)k_mlp, hipFuncAttributeMaxDynamicSharedMemorySize, SMEM_MLP);
  k_mlp<<<1024, NTHREADS, SMEM_MLP, stream>>>(outb, w1T, w2T, b1, b2, out);
}
